// TopKRouter_35287451304121
// MI455X (gfx1250) — matrix operand order repaired and checked by reading, not compiled, not run on hardware
//
#include <hip/hip_runtime.h>

typedef __attribute__((ext_vector_type(16))) _Float16 v16h;
typedef __attribute__((ext_vector_type(8)))  float    v8f;

#define NUM_EXPERTS 64
#define DIM         4096
#define KCHUNK      128
#define NCHUNK      (DIM / KCHUNK)             // 32
#define NSTEP       (KCHUNK / 32)              // 4
#define B_STRIDE    136                        // halves per expert row in LDS (128+8 pad -> 272B)
#define BBUF        (NUM_EXPERTS * B_STRIDE)   // halves per LDS B buffer (8704)
#define LG_STRIDE   66                         // floats per token row in epilogue scratch
#define THREADS     128
#define WAVES       4
#define TOK_PER_WAVE 16
#define TOK_PER_BLOCK (WAVES * TOK_PER_WAVE)   // 64
#define NBLOCKS     (16384 / TOK_PER_BLOCK)    // 256

union SMem {
    _Float16 bh[2 * BBUF];                         // 2 * 17408 B = 34816 B (double-buffered B)
    float    lg[WAVES * TOK_PER_WAVE * LG_STRIDE]; // 16896 B (epilogue logits scratch)
};

union HVec { v16h v; uint4 q[2]; _Float16 h[16]; };

// ---------------- Kernel 1: W fp32 -> f16 into workspace (row-major [64][4096]) ---------------
__global__ __launch_bounds__(256) void convert_w_kernel(const float4* __restrict__ w,
                                                        _Float16* __restrict__ wh) {
    int i = blockIdx.x * blockDim.x + threadIdx.x;   // 0..65535, 4 elems each
    float4 f = w[i];
    union { _Float16 h[4]; uint2 u; } t;
    t.h[0] = (_Float16)f.x; t.h[1] = (_Float16)f.y;
    t.h[2] = (_Float16)f.z; t.h[3] = (_Float16)f.w;
    *(uint2*)(wh + (size_t)i * 4) = t.u;
}

// ---------------- Kernel 2: fused router GEMM (WMMA f16, fp32 acc) + top-2 softmax ------------
// Chunk staging map: 64 experts x 128 halves = 1024 uint4 groups; u = tid + i*128,
// expert = u >> 4 (16 groups per expert row), group-in-row = u & 15.
__global__ __launch_bounds__(THREADS) void router_topk_kernel(const float* __restrict__ X,
                                                              const _Float16* __restrict__ Wh,
                                                              float* __restrict__ out_scores,
                                                              float* __restrict__ out_idx) {
    __shared__ SMem smem;

    const int tid    = threadIdx.x;
    const int wave   = tid >> 5;
    const int lane   = tid & 31;
    const int laneLo = lane & 15;
    const int laneHi = lane >> 4;          // 0 or 1

    const int tokenWave = blockIdx.x * TOK_PER_BLOCK + wave * TOK_PER_WAVE;

    v8f acc0 = {}, acc1 = {}, acc2 = {}, acc3 = {};

    const float* __restrict__ arow = X + (size_t)(tokenWave + laneLo) * DIM;

    // ---- prologue: stage chunk 0 into LDS buffer 0 ----
    {
        uint4 st[8];
        #pragma unroll
        for (int i = 0; i < 8; ++i) {
            int u  = tid + i * THREADS;
            int e  = u >> 4;
            int c8 = u & 15;
            st[i] = *(const uint4*)(Wh + (size_t)e * DIM + c8 * 8);
        }
        #pragma unroll
        for (int i = 0; i < 8; ++i) {
            int u  = tid + i * THREADS;
            int e  = u >> 4;
            int c8 = u & 15;
            *(uint4*)(&smem.bh[e * B_STRIDE + c8 * 8]) = st[i];
        }
    }
    __syncthreads();

    const int kb = laneHi * 8;             // B K-offset within a 32-step for this lane half

    for (int c = 0; c < NCHUNK; ++c) {
        const int k0    = c * KCHUNK;
        const int knext = (k0 + KCHUNK) & (DIM - 1);      // wraps on last chunk (harmless)
        const _Float16* __restrict__ bbuf = smem.bh + (c & 1) * BBUF;
        _Float16* __restrict__ nbuf       = smem.bh + ((c + 1) & 1) * BBUF;

        float4 Af[2][4];
        HVec   Bf[2][4];
        uint4  stage[8];

        // ---- step-0 loads (A from global, B frags from LDS) ----
        {
            int ka = k0 + laneHi * 8;
            Af[0][0] = *(const float4*)(arow + ka);
            Af[0][1] = *(const float4*)(arow + ka + 4);
            Af[0][2] = *(const float4*)(arow + ka + 16);
            Af[0][3] = *(const float4*)(arow + ka + 20);
            const _Float16* bp = &bbuf[laneLo * B_STRIDE + kb];
            #pragma unroll
            for (int t = 0; t < 4; ++t) {
                Bf[0][t].q[0] = *(const uint4*)(bp + t * 16 * B_STRIDE);
                Bf[0][t].q[1] = *(const uint4*)(bp + t * 16 * B_STRIDE + 16);
            }
        }

        #pragma unroll
        for (int s = 0; s < NSTEP; ++s) {
            const int p = s & 1;
            const int q = p ^ 1;

            // ---- issue next-step loads first (distinct regs -> partial waits only) ----
            if (s + 1 < NSTEP) {
                int ka = k0 + (s + 1) * 32 + laneHi * 8;
                Af[q][0] = *(const float4*)(arow + ka);
                Af[q][1] = *(const float4*)(arow + ka + 4);
                Af[q][2] = *(const float4*)(arow + ka + 16);
                Af[q][3] = *(const float4*)(arow + ka + 20);
                // two staging loads per step, issued after this step's A loads so the
                // in-order LOADcnt decrement never stalls the A-consume path on them
                #pragma unroll
                for (int j = 0; j < 2; ++j) {
                    int u  = tid + (2 * s + j) * THREADS;
                    int e  = u >> 4;
                    int c8 = u & 15;
                    stage[2 * s + j] = *(const uint4*)(Wh + (size_t)e * DIM + knext + c8 * 8);
                }
                const _Float16* bp = &bbuf[laneLo * B_STRIDE + (s + 1) * 32 + kb];
                #pragma unroll
                for (int t = 0; t < 4; ++t) {
                    Bf[q][t].q[0] = *(const uint4*)(bp + t * 16 * B_STRIDE);
                    Bf[q][t].q[1] = *(const uint4*)(bp + t * 16 * B_STRIDE + 16);
                }
            } else {
                #pragma unroll
                for (int j = 0; j < 2; ++j) {
                    int u  = tid + (6 + j) * THREADS;
                    int e  = u >> 4;
                    int c8 = u & 15;
                    stage[6 + j] = *(const uint4*)(Wh + (size_t)e * DIM + knext + c8 * 8);
                }
            }

            // ---- convert current A floats -> f16 fragment ----
            HVec A;
            A.h[0]=(_Float16)Af[p][0].x;  A.h[1]=(_Float16)Af[p][0].y;
            A.h[2]=(_Float16)Af[p][0].z;  A.h[3]=(_Float16)Af[p][0].w;
            A.h[4]=(_Float16)Af[p][1].x;  A.h[5]=(_Float16)Af[p][1].y;
            A.h[6]=(_Float16)Af[p][1].z;  A.h[7]=(_Float16)Af[p][1].w;
            A.h[8]=(_Float16)Af[p][2].x;  A.h[9]=(_Float16)Af[p][2].y;
            A.h[10]=(_Float16)Af[p][2].z; A.h[11]=(_Float16)Af[p][2].w;
            A.h[12]=(_Float16)Af[p][3].x; A.h[13]=(_Float16)Af[p][3].y;
            A.h[14]=(_Float16)Af[p][3].z; A.h[15]=(_Float16)Af[p][3].w;

            acc0 = __builtin_amdgcn_wmma_f32_16x16x32_f16(false, A.v, false, Bf[p][0].v, (short)0, acc0, false, false);
            acc1 = __builtin_amdgcn_wmma_f32_16x16x32_f16(false, A.v, false, Bf[p][1].v, (short)0, acc1, false, false);
            acc2 = __builtin_amdgcn_wmma_f32_16x16x32_f16(false, A.v, false, Bf[p][2].v, (short)0, acc2, false, false);
            acc3 = __builtin_amdgcn_wmma_f32_16x16x32_f16(false, A.v, false, Bf[p][3].v, (short)0, acc3, false, false);
        }

        // ---- dump staged next-chunk B to the other LDS buffer, single barrier per chunk ----
        #pragma unroll
        for (int i = 0; i < 8; ++i) {
            int u  = tid + i * THREADS;
            int e  = u >> 4;
            int c8 = u & 15;
            *(uint4*)(&nbuf[e * B_STRIDE + c8 * 8]) = stage[i];
        }
        __syncthreads();
    }

    // ---- epilogue: spill logits tile to LDS (reusing B region), then per-token top-2 softmax ----
    float* mylg = smem.lg + wave * TOK_PER_WAVE * LG_STRIDE;
    #pragma unroll
    for (int r = 0; r < 8; ++r) {
        int m = laneHi * 8 + r;
        mylg[m * LG_STRIDE + laneLo     ] = acc0[r];
        mylg[m * LG_STRIDE + laneLo + 16] = acc1[r];
        mylg[m * LG_STRIDE + laneLo + 32] = acc2[r];
        mylg[m * LG_STRIDE + laneLo + 48] = acc3[r];
    }
    __syncthreads();

    if (lane < 16) {
        const float* row = mylg + lane * LG_STRIDE;
        float v1 = -3.0e38f, v2 = -3.0e38f;
        int   i1 = 0,        i2 = 0;
        #pragma unroll 8
        for (int n = 0; n < NUM_EXPERTS; ++n) {
            float v = row[n];
            if (v > v1)      { v2 = v1; i2 = i1; v1 = v; i1 = n; }
            else if (v > v2) { v2 = v;  i2 = n; }
        }
        float p2 = 1.0f / (1.0f + __expf(v1 - v2));  // softmax over kept top-2, stable (v2 <= v1)
        float p1 = 1.0f - p2;
        int g = tokenWave + lane;
        out_scores[2 * g]     = p1;
        out_scores[2 * g + 1] = p2;
        out_idx[2 * g]        = (float)i1;
        out_idx[2 * g + 1]    = (float)i2;
    }
}

extern "C" void kernel_launch(void* const* d_in, const int* in_sizes, int n_in,
                              void* d_out, int out_size, void* d_ws, size_t ws_size,
                              hipStream_t stream) {
    const float* X = (const float*)d_in[0];      // (4,4096,4096) fp32
    const float* W = (const float*)d_in[1];      // (64,4096) fp32
    _Float16* Wh   = (_Float16*)d_ws;            // 512 KB f16 weights

    float* out        = (float*)d_out;
    float* out_scores = out;                     // (4,4096,2) scores
    float* out_idx    = out + 32768;             // (4,4096,2) indices as floats

    hipLaunchKernelGGL(convert_w_kernel, dim3(256), dim3(256), 0, stream,
                       (const float4*)W, Wh);
    hipLaunchKernelGGL(router_topk_kernel, dim3(NBLOCKS), dim3(THREADS), 0, stream,
                       X, Wh, out_scores, out_idx);
}
